// Program_88124138979590
// MI455X (gfx1250) — matrix operand order repaired and checked by reading, not compiled, not run on hardware
//
#include <hip/hip_runtime.h>

#define BB 524288
#define NSTEP 40
#define H1STRIDE 136   // 128 cols + 8 pad (bf16) -> 272B row stride, 16B aligned, bank-spread
#define W2STRIDE 136

typedef __attribute__((ext_vector_type(16))) __bf16 v16bf;
typedef __attribute__((ext_vector_type(8)))  float  v8f;

// ---------------------------------------------------------------- prologue ---
__global__ __launch_bounds__(256) void convW2K(const float* __restrict__ W2,
                                               __bf16* __restrict__ W2t) {
  int i = blockIdx.x * 256 + threadIdx.x;          // 0..16383
  int k = i >> 7, n = i & 127;
  W2t[n * W2STRIDE + k] = (__bf16)W2[k * 128 + n]; // transpose: [n][k]
}

__global__ __launch_bounds__(256) void initK(const float* __restrict__ in2,
                                             float* __restrict__ x,
                                             float* __restrict__ on) {
  int i = blockIdx.x * 256 + threadIdx.x;
  x[i]  = in2[2 * i + 1];
  on[i] = 0.0f;
}

// ---------------------------------------------------------------- step: MLP ---
// block = 128 threads = 4 waves; each wave does 4 tiles of 16 rows -> 256 rows/block
__global__ __launch_bounds__(128) void mlpStepK(
    const float* __restrict__ in2,
    const float* __restrict__ xCur, const float* __restrict__ onCur,
    const __bf16* __restrict__ gW2t,
    const float* __restrict__ W1, const float* __restrict__ b1,
    const float* __restrict__ b2, const float* __restrict__ W3,
    const float* __restrict__ b3,
    float* __restrict__ xTmp, float* __restrict__ onTmp,
    unsigned char* __restrict__ maskOut, unsigned int* __restrict__ segCounts) {

  __shared__ __align__(16) __bf16 sW2t[128 * W2STRIDE];   // 34816 B
  __shared__ __align__(16) __bf16 sH1[4][16 * H1STRIDE];  // 17408 B
  __shared__ float sW1a[128], sW1b[128], sB1[128], sB2[128], sW3[128];
  __shared__ unsigned int sCnt;

  const int t = threadIdx.x;
  // stage bf16-transposed W2 into LDS (2176 uint4)
  {
    const uint4* g = (const uint4*)gW2t;
    uint4* sh = (uint4*)sW2t;
    for (int i = t; i < (128 * W2STRIDE * 2) / 16; i += 128) sh[i] = g[i];
  }
  if (t < 128) {
    sW1a[t] = W1[t]; sW1b[t] = W1[128 + t];
    sB1[t] = b1[t];  sB2[t] = b2[t]; sW3[t] = W3[t];
  }
  if (t == 0) sCnt = 0;
  __syncthreads();

  const int w = t >> 5, lane = t & 31, half = lane >> 4, m = lane & 15;
  const float b3v = b3[0];
  __bf16* h1w = &sH1[w][0];

  for (int tile = 0; tile < 4; ++tile) {
    const int tileBase = blockIdx.x * 256 + (w * 4 + tile) * 16;
    const int row = tileBase + m;
    const float linv = in2[2 * row];
    const float xv   = xCur[row];

    // ---- layer 1: h1[m][c] = relu(lin*W1[0][c] + x*W1[1][c] + b1[c]), bf16 to LDS
    for (int it = 0; it < 8; ++it) {
      const int c0 = half * 64 + it * 8;
      union { __bf16 h[8]; uint4 u; } pk;
#pragma unroll
      for (int jj = 0; jj < 8; ++jj) {
        const int c = c0 + jj;
        float v = fmaf(linv, sW1a[c], fmaf(xv, sW1b[c], sB1[c]));
        pk.h[jj] = (__bf16)fmaxf(v, 0.0f);
      }
      *(uint4*)&h1w[m * H1STRIDE + c0] = pk.u;
    }
    asm volatile("s_wait_dscnt 0" ::: "memory");  // same-wave LDS RAW

    // ---- A fragments: 16x32 bf16 per K-chunk (ISA layout)
    v16bf A[4];
#pragma unroll
    for (int k = 0; k < 4; ++k) {
      union { uint4 u[2]; v16bf v; } au;
      au.u[0] = *(const uint4*)&h1w[m * H1STRIDE + k * 32 + half * 8];       // K j=0..7
      au.u[1] = *(const uint4*)&h1w[m * H1STRIDE + k * 32 + 16 + half * 8];  // K j=8..15
      A[k] = au.v;
    }

    float p[8];
#pragma unroll
    for (int r = 0; r < 8; ++r) p[r] = 0.0f;

    // ---- layer 2 via WMMA + fold layer 3 dot into epilogue
#pragma unroll
    for (int nt = 0; nt < 8; ++nt) {
      const int col = nt * 16 + m;  // this lane's output column
      v8f acc = {};
#pragma unroll
      for (int k = 0; k < 4; ++k) {
        union { uint4 u[2]; v16bf v; } bu;
        bu.u[0] = *(const uint4*)&sW2t[col * W2STRIDE + k * 32 + half * 8];
        bu.u[1] = *(const uint4*)&sW2t[col * W2STRIDE + k * 32 + 16 + half * 8];   // B k order matches A (table order)
        acc = __builtin_amdgcn_wmma_f32_16x16x32_bf16(
            false, A[k], false, bu.v, (short)0, acc, false, false);
      }
      const float b2v = sB2[col], w3v = sW3[col];
#pragma unroll
      for (int r = 0; r < 8; ++r) {
        float h2 = fmaxf(acc[r] + b2v, 0.0f);  // row = half*8 + r
        p[r] = fmaf(h2, w3v, p[r]);
      }
    }

    // ---- butterfly reduce p[r] over the 16 lanes of each half
#pragma unroll
    for (int d = 1; d < 16; d <<= 1) {
#pragma unroll
      for (int r = 0; r < 8; ++r) {
        int src = ((lane ^ d) << 2);
        p[r] += __uint_as_float(
            __builtin_amdgcn_ds_bpermute(src, __float_as_uint(p[r])));
      }
    }

    // ---- finalize: lanes {0..7,16..23} each own one of the 16 rows
    if (m < 8) {
      const int frow = tileBase + half * 8 + m;
      const float plant = (p[m] + b3v) * 0.01f;
      const float xo = xCur[frow];
      const float ov = onCur[frow];
      const bool  isOnB = ov > 0.5f;
      const float offx = xo - plant;
      const float onx  = offx + 10.0f;
      const float newx = isOnB ? onx : offx;
      const float newOn = isOnB ? (onx > 78.0f ? 0.0f : ov)
                                : (offx <= 66.0f ? 1.0f : ov);
      xTmp[frow]  = newx;
      onTmp[frow] = newOn;
      maskOut[frow] = isOnB ? (unsigned char)1 : (unsigned char)0;
      unsigned bal = __builtin_amdgcn_ballot_w32(!isOnB);  // off-count of 16 rows
      if (lane == 0) atomicAdd(&sCnt, __popc(bal));
    }
  }
  __syncthreads();
  if (t == 0) segCounts[blockIdx.x] = sCnt;
}

// ---------------------------------------------------------------- step: scan ---
__global__ __launch_bounds__(1024) void scanK(const unsigned int* __restrict__ counts,
                                              unsigned int* __restrict__ prefix,
                                              unsigned int* __restrict__ totalOff) {
  __shared__ unsigned int bufA[2048], bufB[2048];
  const int t = threadIdx.x;
  bufA[t] = counts[t];
  bufA[t + 1024] = counts[t + 1024];
  __syncthreads();
  unsigned int* src = bufA;
  unsigned int* dst = bufB;
  for (int off = 1; off < 2048; off <<= 1) {
    for (int i = t; i < 2048; i += 1024) {
      unsigned int v = src[i];
      if (i >= off) v += src[i - off];
      dst[i] = v;
    }
    __syncthreads();
    unsigned int* tmp = src; src = dst; dst = tmp;
  }
  for (int i = t; i < 2048; i += 1024) prefix[i] = src[i] - counts[i];  // exclusive
  if (t == 0) *totalOff = src[2047];
}

// ---------------------------------------------------------------- step: scatter ---
__global__ __launch_bounds__(256) void scatterK(
    const unsigned char* __restrict__ mask,
    const float* __restrict__ xT, const float* __restrict__ onT,
    const unsigned int* __restrict__ prefix,
    const unsigned int* __restrict__ totalOff,
    float* __restrict__ xD, float* __restrict__ onD) {
  const int s = blockIdx.x, t = threadIdx.x;
  const unsigned int i = (unsigned)s * 256u + (unsigned)t;
  const int lane = t & 31, w = t >> 5;
  const bool on = mask[i] != 0;
  const unsigned bal = __builtin_amdgcn_ballot_w32(!on);
  const unsigned rank = __popc(bal & ((1u << lane) - 1u));
  __shared__ unsigned int wcnt[8];
  if (lane == 0) wcnt[w] = __popc(bal);
  __syncthreads();
  unsigned base = 0;
#pragma unroll
  for (int k = 0; k < 8; ++k) if (k < w) base += wcnt[k];
  const unsigned offRank = base + rank;           // off-rank within segment
  const unsigned segPre = prefix[s];
  const unsigned dest = on ? (*totalOff + i - segPre - offRank)
                           : (segPre + offRank);
  xD[dest]  = xT[i];
  onD[dest] = onT[i];
}

// ---------------------------------------------------------------- launch ---
extern "C" void kernel_launch(void* const* d_in, const int* in_sizes, int n_in,
                              void* d_out, int out_size, void* d_ws, size_t ws_size,
                              hipStream_t stream) {
  (void)in_sizes; (void)n_in; (void)out_size; (void)ws_size;
  const float* in2 = (const float*)d_in[0];
  const float* W1  = (const float*)d_in[1];
  const float* b1  = (const float*)d_in[2];
  const float* W2  = (const float*)d_in[3];
  const float* b2  = (const float*)d_in[4];
  const float* W3  = (const float*)d_in[5];
  const float* b3  = (const float*)d_in[6];

  char* ws = (char*)d_ws;
  size_t off = 0;
  auto carve = [&](size_t bytes) -> void* {
    void* p = ws + off;
    off += (bytes + 255) & ~(size_t)255;
    return p;
  };
  float* xA  = (float*)carve((size_t)BB * 4);
  float* xB  = (float*)carve((size_t)BB * 4);
  float* xT  = (float*)carve((size_t)BB * 4);
  float* onA = (float*)carve((size_t)BB * 4);
  float* onB = (float*)carve((size_t)BB * 4);
  float* onT = (float*)carve((size_t)BB * 4);
  unsigned char* maskB   = (unsigned char*)carve((size_t)BB);
  unsigned int* segCnt   = (unsigned int*)carve(2048 * 4);
  unsigned int* segPre   = (unsigned int*)carve(2048 * 4);
  unsigned int* totalOff = (unsigned int*)carve(256);
  __bf16* gW2t           = (__bf16*)carve((size_t)128 * W2STRIDE * 2);

  convW2K<<<64, 256, 0, stream>>>(W2, gW2t);
  initK<<<BB / 256, 256, 0, stream>>>(in2, xA, onA);

  for (int s = 0; s < NSTEP; ++s) {
    float* xc = (s & 1) ? xB : xA;
    float* oc = (s & 1) ? onB : onA;
    float* xn = (s & 1) ? xA : xB;
    float* on = (s & 1) ? onA : onB;
    if (s == NSTEP - 1) xn = (float*)d_out;  // final x lands directly in d_out

    mlpStepK<<<BB / 256, 128, 0, stream>>>(in2, xc, oc, gW2t, W1, b1, b2, W3, b3,
                                           xT, onT, maskB, segCnt);
    scanK<<<1, 1024, 0, stream>>>(segCnt, segPre, totalOff);
    scatterK<<<BB / 256, 256, 0, stream>>>(maskB, xT, onT, segPre, totalOff, xn, on);
  }
}
